// LSTMEncoder_24842090840127
// MI455X (gfx1250) — compile-verified, matrix operand order checked by reading, not run on hardware
//
#include <hip/hip_runtime.h>
#include <hip/hip_bf16.h>

// ---------------------------------------------------------------------------
// LSTM encoder for MI455X (gfx1250, wave32, WMMA + TDM + async-to-LDS).
// Sizes fixed by the reference: B=16, T=1024, IN=8, H=512, L=4, FC 512->256->1
// ---------------------------------------------------------------------------

#define Bsz 16
#define Tsz 1024
#define Hsz 512
#define G4H 2048          // 4*H
#define Msz (Tsz * Bsz)   // 16384 rows, row m = t*16 + b

typedef __attribute__((ext_vector_type(16))) __bf16 v16bf;
typedef __attribute__((ext_vector_type(8)))  float  v8f;

typedef unsigned int u32x4 __attribute__((ext_vector_type(4)));
typedef int          i32x8 __attribute__((ext_vector_type(8)));
typedef int          i32x4 __attribute__((ext_vector_type(4)));

#if defined(__has_builtin)
#  if __has_builtin(__builtin_amdgcn_tensor_load_to_lds)
#    define HAVE_TDM 1
#  endif
#  if __has_builtin(__builtin_amdgcn_s_wait_tensorcnt)
#    define HAVE_WAIT_TENSOR 1
#  endif
#  if __has_builtin(__builtin_amdgcn_s_wait_asynccnt)
#    define HAVE_WAIT_ASYNC 1
#  endif
#endif

struct Frag16 { union { v16bf v; unsigned u[8]; }; };

// bf16 round-to-nearest-even from f32
__device__ inline unsigned short f2bf(float x) {
    union { float f; unsigned u; } v; v.f = x;
    unsigned r = v.u + 0x7FFFu + ((v.u >> 16) & 1u);
    return (unsigned short)(r >> 16);
}

// LDS byte address of a generic pointer to __shared__ memory.
// Per the CDNA5 flat-aperture mapping, the LDS address is the low 32 bits of the flat address.
__device__ inline unsigned lds_offset(const void* p) {
    return (unsigned)(unsigned long long)p;
}

// Load one 16-element bf16 fragment for this lane, row-major source.
// Element->K mapping follows the CDNA5 16-bit A/B VGPR layout:
//   lane<16 holds K in {0..7, 16..23}, lane>=16 holds K in {8..15, 24..31}
__device__ inline void load_frag(Frag16& f, const unsigned short* rowptr, int hi) {
#pragma unroll
    for (int d = 0; d < 8; ++d) {
        int kb = ((d < 4) ? 0 : 16) + hi * 8 + (d & 3) * 2;
        f.u[d] = *reinterpret_cast<const unsigned*>(rowptr + kb);
    }
}

__device__ inline v8f wmma_bf16(const Frag16& a, const Frag16& b, v8f c) {
    return __builtin_amdgcn_wmma_f32_16x16x32_bf16(
        false, a.v, false, b.v, (short)0, c, false, false);
}

__device__ inline float sigmoidf_(float x) {
    x = fminf(fmaxf(x, -30.f), 30.f);
    return 1.f / (1.f + __expf(-x));
}
__device__ inline float tanhf_(float x) {
    x = fminf(fmaxf(x, -15.f), 15.f);
    float e = __expf(2.f * x);
    return (e - 1.f) / (e + 1.f);
}

// ---- CDNA5 async/TDM primitives ------------------------------------------

__device__ inline void async_load_b128(unsigned lds_off, unsigned long long gaddr) {
    asm volatile("global_load_async_to_lds_b128 %0, %1, off"
                 :: "v"(lds_off), "v"(gaddr) : "memory");
}

__device__ inline void wait_asynccnt0() {
#if defined(HAVE_WAIT_ASYNC)
    __builtin_amdgcn_s_wait_asynccnt(0);
#else
    asm volatile("s_wait_asynccnt 0" ::: "memory");
#endif
}

__device__ inline void wait_tensorcnt(int n) {
#if defined(HAVE_WAIT_TENSOR)
    if (n == 0) __builtin_amdgcn_s_wait_tensorcnt((short)0);
    else        __builtin_amdgcn_s_wait_tensorcnt((short)1);
#else
    if (n == 0) asm volatile("s_wait_tensorcnt 0" ::: "memory");
    else        asm volatile("s_wait_tensorcnt 1" ::: "memory");
#endif
}

// Issue a TDM load of a 16x16 f32 tile (row stride 2048 floats) into LDS.
// D# group0: count=1, lds_addr, 57-bit global_addr, type=2 ("image").
// D# group1: data_size=4B, tensor=16x16, tile=16x16, dim0_stride=2048.
// This toolchain exposes the 6-arg builtin:
//   (uint32x4 g0, int32x8 g1, int32x4 g2, int32x4 g3, int32x8 g4, i32 cpol)
__device__ inline void tdm_prefetch_tile16x16_f32(const float* gsrc, float* ldst) {
#if defined(HAVE_TDM)
    unsigned long long ga = (unsigned long long)gsrc;
    unsigned lds = lds_offset(ldst);
    u32x4 g0;
    g0[0] = (unsigned)__builtin_amdgcn_readfirstlane(1u);            // count=1
    g0[1] = (unsigned)__builtin_amdgcn_readfirstlane(lds);           // lds_addr
    g0[2] = (unsigned)__builtin_amdgcn_readfirstlane((unsigned)ga);  // gaddr lo
    g0[3] = (unsigned)__builtin_amdgcn_readfirstlane(
                ((unsigned)(ga >> 32) & 0x01FFFFFFu) | 0x80000000u); // gaddr hi | type=2
    i32x8 g1;
    g1[0] = __builtin_amdgcn_readfirstlane(0x20000);    // data_size=2 (4 bytes)
    g1[1] = __builtin_amdgcn_readfirstlane(16 << 16);   // tensor_dim0=16 (lo16)
    g1[2] = __builtin_amdgcn_readfirstlane(16 << 16);   // dim0 hi=0 | tensor_dim1=16 lo16
    g1[3] = __builtin_amdgcn_readfirstlane(16 << 16);   // dim1 hi=0 | tile_dim0=16
    g1[4] = __builtin_amdgcn_readfirstlane(16);         // tile_dim1=16 | tile_dim2=0
    g1[5] = __builtin_amdgcn_readfirstlane(2048);       // tensor_dim0_stride=2048
    g1[6] = 0;
    g1[7] = 0;
    i32x4 g2 = {0, 0, 0, 0};
    i32x4 g3 = {0, 0, 0, 0};
    i32x8 g4 = {0, 0, 0, 0, 0, 0, 0, 0};
    __builtin_amdgcn_tensor_load_to_lds(g0, g1, g2, g3, g4, 0);
#else
    (void)gsrc; (void)ldst;
#endif
}

// ---------------------------------------------------------------------------
// f32 -> bf16 conversion (grid-stride)
// ---------------------------------------------------------------------------
__global__ void k_f32_to_bf16(const float* __restrict__ in,
                              unsigned short* __restrict__ out, size_t n) {
    size_t i = (size_t)blockIdx.x * blockDim.x + threadIdx.x;
    size_t stride = (size_t)gridDim.x * blockDim.x;
    for (; i < n; i += stride) out[i] = f2bf(in[i]);
}

// ---------------------------------------------------------------------------
// Upscale: x(B,T,8) @ W_up(H,8)^T + b_up -> X0 bf16, rows m = t*16+b, cols H
// ---------------------------------------------------------------------------
__global__ void k_upscale(const float* __restrict__ x,
                          const float* __restrict__ Wup,
                          const float* __restrict__ bup,
                          unsigned short* __restrict__ X0) {
    size_t idx = (size_t)blockIdx.x * blockDim.x + threadIdx.x;
    if (idx >= (size_t)Msz * Hsz) return;
    int m = (int)(idx >> 9), n = (int)(idx & 511);
    int t = m >> 4, b = m & 15;
    const float* xr = x + ((size_t)b * Tsz + t) * 8;
    const float* wr = Wup + (size_t)n * 8;
    float s = bup[n];
#pragma unroll
    for (int k = 0; k < 8; ++k) s += xr[k] * wr[k];
    X0[idx] = f2bf(s);
}

// ---------------------------------------------------------------------------
// WMMA GEMM: out(M x N, f32) = A(M x K, bf16) @ W(N x K, bf16)^T + bias1+bias2
// grid.x = M/32, grid.y = N/64, one wave (32 threads) per block.
// Each wave: 32x64 output tile (2 A-frags x 4 B-frags, 8 accumulators),
// K-loop in steps of 32; every B fragment feeds two WMMAs.
// ---------------------------------------------------------------------------
__global__ void k_wmma_gemm_bias(const unsigned short* __restrict__ A,
                                 const unsigned short* __restrict__ W,
                                 const float* __restrict__ bias1,
                                 const float* __restrict__ bias2,
                                 float* __restrict__ out,
                                 int lda, int ldb, int ldo, int K, int relu) {
    int lane = threadIdx.x & 31;
    int hi = lane >> 4, lo = lane & 15;
    int m0 = blockIdx.x * 32, n0 = blockIdx.y * 64;
    v8f acc[2][4] = {};
    for (int k0 = 0; k0 < K; k0 += 32) {
        Frag16 a0, a1;
        load_frag(a0, A + (size_t)(m0 + lo) * lda + k0, hi);
        load_frag(a1, A + (size_t)(m0 + 16 + lo) * lda + k0, hi);
#pragma unroll
        for (int j = 0; j < 4; ++j) {
            Frag16 b;
            load_frag(b, W + (size_t)(n0 + j * 16 + lo) * ldb + k0, hi);
            acc[0][j] = wmma_bf16(a0, b, acc[0][j]);
            acc[1][j] = wmma_bf16(a1, b, acc[1][j]);
        }
    }
#pragma unroll
    for (int i = 0; i < 2; ++i) {
#pragma unroll
        for (int j = 0; j < 4; ++j) {
            int col = n0 + j * 16 + lo;
            float bb = (bias1 ? bias1[col] : 0.f) + (bias2 ? bias2[col] : 0.f);
#pragma unroll
            for (int r = 0; r < 8; ++r) {
                float v = acc[i][j][r] + bb;
                if (relu) v = fmaxf(v, 0.f);
                out[(size_t)(m0 + i * 16 + r + 8 * hi) * ldo + col] = v;
            }
        }
    }
}

// ---------------------------------------------------------------------------
// LSTM recurrence for one layer.
//   pre  : (T*16) x 2048 f32, gate pre-activations incl. both biases
//   Whh  : 2048 x 512 bf16 (row = gate output, col = h input)
//   Xout : (T*16) x 512 bf16 -- h_t written here; h_{t-1} read from here
// Grid: 16 blocks x 256 threads (8 waves). Block owns h-tiles {2*blk, 2*blk+1}.
// Wave w: tile tau=w>>2, gate=w&3.
//   * W_hh panel (16 frags = 128 VGPRs) preloaded into registers, reused 1024x
//   * h_{t-1} panel staged once per step into LDS via async-to-LDS loads
//   * per-wave 16x16 pre tile prefetched one step ahead via TDM (double buffer)
//   * cell state c lives in the gate-0 wave's registers
//   * steps separated by an atomic grid barrier with device fence
// ---------------------------------------------------------------------------
__global__ void k_lstm_recurrence(const float* __restrict__ pre,
                                  const unsigned short* __restrict__ Whh,
                                  unsigned short* __restrict__ Xout,
                                  unsigned* sync_cnt, unsigned* sync_gen) {
    __shared__ float exch[8][256];                  // gate-tile exchange, 8 KB
    __shared__ unsigned short ldsA[16 * Hsz];       // h_{t-1} staging, 16 KB
#if defined(HAVE_TDM)
    __shared__ float ldsPre[2][8][256];             // TDM double buffer, 16 KB
#endif

    int tid = threadIdx.x;
    int w = tid >> 5, lane = tid & 31;
    int hi = lane >> 4, lo = lane & 15;
    int tau = w >> 2, gate = w & 3;
    int j = blockIdx.x * 2 + tau;            // global h-column tile (0..31)
    int gc = gate * Hsz + j * 16;            // gate column base in 4H

    // Preload this wave's W_hh B-panel: rows [gc, gc+16), K = 0..511
    Frag16 Bf[16];
    {
        const unsigned short* brow = Whh + (size_t)(gc + lo) * Hsz;
#pragma unroll
        for (int s = 0; s < 16; ++s) load_frag(Bf[s], brow + s * 32, hi);
    }

#if defined(HAVE_TDM)
    // prefetch the t=0 pre tile
    tdm_prefetch_tile16x16_f32(pre + gc, &ldsPre[0][w][0]);
#endif

    v8f c = {};                              // persistent cell state (gate==0)
    const bool combiner = (gate == 0);
    volatile unsigned* genv = (volatile unsigned*)sync_gen;
    volatile unsigned* cntv = (volatile unsigned*)sync_cnt;

    for (int t = 0; t < Tsz; ++t) {
        // ---- cooperative async stage of h_{t-1} (16 KB) into LDS ----------
        if (t > 0) {
            const unsigned short* src = Xout + (size_t)(t - 1) * 16 * Hsz;
#pragma unroll
            for (int it = 0; it < 4; ++it) {
                int i = (it * 256 + tid) * 8;          // 8 halves = 16 bytes
                async_load_b128(lds_offset(&ldsA[i]),
                                (unsigned long long)(src + i));
            }
            wait_asynccnt0();
        }
        __syncthreads();                     // panel visible to all 8 waves

#if defined(HAVE_TDM)
        // issue TDM prefetch of next step's pre tile while computing this one
        if (t + 1 < Tsz)
            tdm_prefetch_tile16x16_f32(pre + (size_t)(t + 1) * 16 * G4H + gc,
                                       &ldsPre[(t + 1) & 1][w][0]);
#endif

        v8f acc = {};
        if (t > 0) {
#pragma unroll
            for (int s = 0; s < 16; ++s) {
                Frag16 a;
                load_frag(a, &ldsA[(size_t)lo * Hsz + s * 32], hi);
                acc = wmma_bf16(a, Bf[s], acc);
            }
        }

        // ---- add input-side gate pre-activations --------------------------
#if defined(HAVE_TDM)
        wait_tensorcnt((t + 1 < Tsz) ? 1 : 0);
        const float* ptile = &ldsPre[t & 1][w][0];
#pragma unroll
        for (int r = 0; r < 8; ++r) acc[r] += ptile[(r + 8 * hi) * 16 + lo];
#else
        const float* prow = pre + (size_t)(t * 16) * G4H + gc + lo;
#pragma unroll
        for (int r = 0; r < 8; ++r) acc[r] += prow[(size_t)(r + 8 * hi) * G4H];
#endif

        // publish gate tile to LDS
#pragma unroll
        for (int r = 0; r < 8; ++r) exch[w][(r + 8 * hi) * 16 + lo] = acc[r];
        __syncthreads();

        if (combiner) {
            int wi = w;   // = 4*tau ; gates i,f,g,o at wi+0..3
#pragma unroll
            for (int r = 0; r < 8; ++r) {
                int idx = (r + 8 * hi) * 16 + lo;
                float iv = sigmoidf_(exch[wi + 0][idx]);
                float fv = sigmoidf_(exch[wi + 1][idx]);
                float gv = tanhf_  (exch[wi + 2][idx]);
                float ov = sigmoidf_(exch[wi + 3][idx]);
                float cn = fv * c[r] + iv * gv;
                c[r] = cn;
                float h = ov * tanhf_(cn);
                Xout[(size_t)(t * 16 + r + 8 * hi) * Hsz + j * 16 + lo] = f2bf(h);
            }
        }
        __syncthreads();

        // ---- grid barrier across the 16 blocks ----------------------------
        if (tid == 0) {
            __threadfence();
            unsigned g = *genv;
            if (atomicAdd(sync_cnt, 1u) == gridDim.x - 1) {
                *cntv = 0u;
                __threadfence();
                atomicAdd(sync_gen, 1u);
            } else {
                while (*genv == g) __builtin_amdgcn_s_sleep(2);
            }
            __threadfence();
        }
        __syncthreads();
    }
}

// ---------------------------------------------------------------------------
// FC2: out[b*T + t] = relu( dot(fc1[m, 0:256], W_fc2) + b_fc2 ),  m = t*16+b
// One wave per row, lane-strided dot + xor-shuffle reduction.
// ---------------------------------------------------------------------------
__global__ void k_fc2(const float* __restrict__ fc1,
                      const float* __restrict__ w2,
                      const float* __restrict__ b2,
                      float* __restrict__ out) {
    int tid = threadIdx.x, lane = tid & 31, wv = tid >> 5;
    int m = blockIdx.x * 8 + wv;
    const float* row = fc1 + (size_t)m * 256;
    float s = 0.f;
#pragma unroll
    for (int i = 0; i < 8; ++i) {
        int k = lane + 32 * i;
        s += row[k] * w2[k];
    }
#pragma unroll
    for (int off = 16; off > 0; off >>= 1) s += __shfl_xor(s, off, 32);
    if (lane == 0) {
        int t = m >> 4, b = m & 15;
        out[(size_t)b * Tsz + t] = fmaxf(s + b2[0], 0.f);
    }
}

// ---------------------------------------------------------------------------
extern "C" void kernel_launch(void* const* d_in, const int* in_sizes, int n_in,
                              void* d_out, int out_size, void* d_ws, size_t ws_size,
                              hipStream_t stream) {
    const float* x     = (const float*)d_in[0];
    const float* W_up  = (const float*)d_in[1];
    const float* b_up  = (const float*)d_in[2];
    const float* W_ih  = (const float*)d_in[3];   // (4, 2048, 512)
    const float* W_hh  = (const float*)d_in[4];   // (4, 2048, 512)
    const float* b_ih  = (const float*)d_in[5];   // (4, 2048)
    const float* b_hh  = (const float*)d_in[6];   // (4, 2048)
    const float* W_fc1 = (const float*)d_in[7];   // (256, 512)
    const float* b_fc1 = (const float*)d_in[8];
    const float* W_fc2 = (const float*)d_in[9];   // (1, 256)
    const float* b_fc2 = (const float*)d_in[10];
    (void)in_sizes; (void)n_in; (void)out_size; (void)ws_size;

    char* ws = (char*)d_ws;
    size_t off = 0;
    auto alloc = [&](size_t bytes) -> void* {
        void* p = ws + off;
        off = (off + bytes + 255) & ~(size_t)255;
        return p;
    };
    const size_t WNEL = (size_t)4 * G4H * Hsz;
    unsigned short* Wih_bf  = (unsigned short*)alloc(WNEL * 2);
    unsigned short* Whh_bf  = (unsigned short*)alloc(WNEL * 2);
    unsigned short* Wfc1_bf = (unsigned short*)alloc((size_t)256 * Hsz * 2);
    unsigned short* X_a     = (unsigned short*)alloc((size_t)Msz * Hsz * 2);
    unsigned short* X_b     = (unsigned short*)alloc((size_t)Msz * Hsz * 2);
    float*          pre     = (float*)alloc((size_t)Msz * G4H * 4);   // 128 MB
    float*          fc1out  = (float*)alloc((size_t)Msz * 256 * 4);
    unsigned*       sync    = (unsigned*)alloc(256);
    unsigned* sync_cnt = sync;
    unsigned* sync_gen = sync + 1;

    (void)hipMemsetAsync(sync, 0, 256, stream);

    // weight precision conversion (f32 -> bf16)
    k_f32_to_bf16<<<2048, 256, 0, stream>>>(W_ih,  Wih_bf,  WNEL);
    k_f32_to_bf16<<<2048, 256, 0, stream>>>(W_hh,  Whh_bf,  WNEL);
    k_f32_to_bf16<<<256,  256, 0, stream>>>(W_fc1, Wfc1_bf, (size_t)256 * Hsz);

    // upscale 8 -> 512
    k_upscale<<<(Msz * Hsz) / 256, 256, 0, stream>>>(x, W_up, b_up, X_a);

    unsigned short* Xin = X_a;
    unsigned short* Xout = X_b;
    for (int l = 0; l < 4; ++l) {
        // pre = Xin @ W_ih[l]^T + b_ih[l] + b_hh[l]   (16384 x 2048)
        dim3 g1(Msz / 32, G4H / 64);
        k_wmma_gemm_bias<<<g1, 32, 0, stream>>>(
            Xin, Wih_bf + (size_t)l * G4H * Hsz,
            b_ih + (size_t)l * G4H, b_hh + (size_t)l * G4H,
            pre, Hsz, Hsz, G4H, Hsz, 0);
        // sequential recurrence over T; writes layer output (= h sequence)
        k_lstm_recurrence<<<16, 256, 0, stream>>>(
            pre, Whh_bf + (size_t)l * G4H * Hsz, Xout, sync_cnt, sync_gen);
        unsigned short* tmp = Xin; Xin = Xout; Xout = tmp;
    }

    // FC1 + ReLU: 16384 x 256
    dim3 g2(Msz / 32, 256 / 64);
    k_wmma_gemm_bias<<<g2, 32, 0, stream>>>(
        Xin, Wfc1_bf, b_fc1, nullptr, fc1out, Hsz, Hsz, 256, Hsz, 1);

    // FC2 + ReLU -> d_out (B, T, 1) f32
    k_fc2<<<Msz / 8, 256, 0, stream>>>(fc1out, W_fc2, b_fc2, (float*)d_out);
}
